// SparseRouter_90151363543476
// MI455X (gfx1250) — matrix operand order repaired and checked by reading, not compiled, not run on hardware
//
#include <hip/hip_runtime.h>
#include <hip/hip_bf16.h>
#include <math.h>

// ---------------- problem constants ----------------
#define HID   1024          // hidden
#define FF    2048          // 2*hidden
#define NEXP  8
#define TOPK  2
#define NTOK  8192          // 4*2048 tokens
#define MT    32            // tokens per FFN workgroup tile
#define A_STR 1032          // LDS row stride (bf16 elems) for A tile (pad -> conflict-free)
#define H_STR 2056          // LDS row stride (bf16 elems) for H tile

typedef __bf16 bf16;
typedef __attribute__((ext_vector_type(16))) bf16  v16bf;
typedef __attribute__((ext_vector_type(8)))  bf16  bf16x8;
typedef __attribute__((ext_vector_type(8)))  float v8f;

union FragU { bf16x8 h[2]; v16bf v; };

__device__ __forceinline__ v8f wmma_bf16(v16bf a, v16bf b, v8f c) {
    // D = A(16x32 bf16) * B(32x16 bf16) + C(16x16 f32)
    return __builtin_amdgcn_wmma_f32_16x16x32_bf16(
        /*neg_a*/false, a, /*neg_b*/false, b,
        /*c_mod*/(short)0, c, /*reuse_a*/false, /*reuse_b*/false);
}

__device__ __forceinline__ float gelu_exact(float x) {
    return 0.5f * x * (1.0f + erff(x * 0.70710678118654752f));
}

// ---------------- kernel 1: router logits + top2 + softmax ----------------
__global__ __launch_bounds__(256) void router_kernel(
    const float* __restrict__ x, const float* __restrict__ rw,
    int* __restrict__ tidx, float* __restrict__ tw)
{
    __shared__ float rws[NEXP * HID];
    for (int i = threadIdx.x; i < NEXP * HID; i += 256) rws[i] = rw[i];
    __syncthreads();

    const int lane = threadIdx.x & 31;
    const int wave = threadIdx.x >> 5;
    const int t = blockIdx.x * 8 + wave;

    const float* xr = x + (size_t)t * HID;
    float acc[NEXP];
#pragma unroll
    for (int e = 0; e < NEXP; ++e) acc[e] = 0.f;
    for (int k = lane; k < HID; k += 32) {
        float xv = xr[k];
#pragma unroll
        for (int e = 0; e < NEXP; ++e) acc[e] += xv * rws[e * HID + k];
    }
#pragma unroll
    for (int e = 0; e < NEXP; ++e)
        for (int off = 16; off; off >>= 1) acc[e] += __shfl_xor(acc[e], off, 32);

    if (lane == 0) {
        int b0 = 0; float v0 = acc[0];
        for (int e = 1; e < NEXP; ++e) if (acc[e] > v0) { v0 = acc[e]; b0 = e; }
        int b1 = -1; float v1 = -1e30f;
        for (int e = 0; e < NEXP; ++e) if (e != b0 && acc[e] > v1) { v1 = acc[e]; b1 = e; }
        float e1 = expf(v1 - v0);      // softmax over [v0, v1], v0 is max
        float inv = 1.0f / (1.0f + e1);
        tidx[t * 2 + 0] = b0; tidx[t * 2 + 1] = b1;
        tw[t * 2 + 0] = inv;  tw[t * 2 + 1] = e1 * inv;
    }
}

// ---------------- kernel 2: x fp32 -> bf16 ----------------
__global__ __launch_bounds__(256) void cvt_x_kernel(
    const float* __restrict__ x, bf16* __restrict__ xb, int n)
{
    int i = blockIdx.x * 256 + threadIdx.x;
    if (i < n) xb[i] = (bf16)x[i];
}

// ---------------- kernel 3: weight [E][R][C] fp32 -> transposed bf16 [E][C][R] ----------------
__global__ __launch_bounds__(256) void transpose_cvt_kernel(
    const float* __restrict__ in, bf16* __restrict__ out, int R, int C)
{
    __shared__ float tile[32][33];
    const int e  = blockIdx.z;
    const int r0 = blockIdx.y * 32, c0 = blockIdx.x * 32;
    const int tx = threadIdx.x & 31, ty = threadIdx.x >> 5;   // 32 x 8
    const float* src = in  + (size_t)e * R * C;
    bf16*        dst = out + (size_t)e * R * C;
#pragma unroll
    for (int i = 0; i < 4; ++i)
        tile[ty + 8 * i][tx] = src[(size_t)(r0 + ty + 8 * i) * C + c0 + tx];
    __syncthreads();
#pragma unroll
    for (int i = 0; i < 4; ++i)
        dst[(size_t)(c0 + ty + 8 * i) * R + r0 + tx] = (bf16)tile[tx][ty + 8 * i];
}

// ---------------- kernel 4: counts=0, then scatter assignments by expert ----------------
__global__ void zero_counts_kernel(int* counts) {
    if (threadIdx.x < NEXP) counts[threadIdx.x] = 0;
}
__global__ __launch_bounds__(256) void scatter_kernel(
    const int* __restrict__ tidx, int* __restrict__ counts, int* __restrict__ lists)
{
    int t = blockIdx.x * 256 + threadIdx.x;
    if (t < NTOK) {
#pragma unroll
        for (int k = 0; k < TOPK; ++k) {
            int e = tidx[t * 2 + k];
            int slot = atomicAdd(&counts[e], 1);          // slot order irrelevant to math
            lists[e * NTOK + slot] = t * 2 + k;           // assignment id
        }
    }
}

// ---------------- kernel 5: fused expert FFN with bf16 WMMA ----------------
// grid = NEXP * 256 tiles; block = 256 threads = 8 waves; M-tile = 32 tokens.
// Each wave iteration handles an N-PAIR (two adjacent 16-wide N tiles) so the
// A fragments (ds_load_b128) are reused by two B tiles -> 4 WMMAs per K-step.
// K-loops are #pragma unroll 1 to keep ds_loads in-loop (prevents the LICM
// register blow-up / scratch spilling observed with full unroll).
__global__ __launch_bounds__(256) void moe_ffn_kernel(
    const bf16* __restrict__ xb,   // [NTOK][HID] bf16
    const bf16* __restrict__ w1t,  // [E][FF][HID] bf16 (transposed: N-major)
    const bf16* __restrict__ w2t,  // [E][HID][FF] bf16 (transposed: N-major)
    const int*  __restrict__ counts,
    const int*  __restrict__ lists,
    float*      __restrict__ ok)   // [NTOK*2][HID] fp32, indexed by assignment id
{
    __shared__ bf16 Als[MT * A_STR];   // 66 KB
    __shared__ bf16 Hls[MT * H_STR];   // 132 KB
    __shared__ int  aid[MT];

    const int e    = blockIdx.x >> 8;
    const int tile = blockIdx.x & 255;
    const int cnt  = counts[e];
    const int base = tile * MT;
    if (base >= cnt) return;

    const int lane = threadIdx.x & 31;
    const int wave = threadIdx.x >> 5;
    const int half = lane >> 4;        // K-half for WMMA fragment layout
    const int nlo  = lane & 15;        // N / M sub-index

    if (threadIdx.x < MT) {
        int idx = base + threadIdx.x;
        aid[threadIdx.x] = lists[e * NTOK + ((idx < cnt) ? idx : 0)]; // clamp pad rows
    }
    __syncthreads();

    // ---- stage A tile (gathered token rows) into LDS, 32-bit granules ----
    {
        const unsigned int* xbu = (const unsigned int*)xb;
        unsigned int* Au = (unsigned int*)Als;
#pragma unroll 1
        for (int i = threadIdx.x; i < MT * (HID / 2); i += 256) {
            int row = i >> 9;            // / 512 uints per row
            int cp  = i & 511;
            int tok = aid[row] >> 1;
            Au[row * (A_STR / 2) + cp] = xbu[(size_t)tok * (HID / 2) + cp];
        }
    }
    __syncthreads();

    // ---- phase 1: A @ W1 -> gelu -> Hls ----
    const bf16* w1e = w1t + (size_t)e * FF * HID;
    const bf16* ar0 = &Als[nlo * A_STR];
    const bf16* ar1 = &Als[(16 + nlo) * A_STR];
#pragma unroll 1
    for (int np = wave; np < FF / 32; np += 8) {       // 64 N-pairs, 8 per wave
        const int n0 = np * 32;
        v8f acc00 = {0,0,0,0,0,0,0,0}, acc01 = {0,0,0,0,0,0,0,0};
        v8f acc10 = {0,0,0,0,0,0,0,0}, acc11 = {0,0,0,0,0,0,0,0};
        const bf16* brow0 = w1e + (size_t)(n0 + nlo) * HID + 16 * half;
        const bf16* brow1 = brow0 + (size_t)16 * HID;
#pragma unroll 1
        for (int k0 = 0; k0 < HID; k0 += 32) {
            FragU fa0, fa1, fb0, fb1;
            fa0.h[0] = *(const bf16x8*)(ar0 + k0 +     16 * half);
            fa0.h[1] = *(const bf16x8*)(ar0 + k0 + 8 + 16 * half);
            fa1.h[0] = *(const bf16x8*)(ar1 + k0 +     16 * half);
            fa1.h[1] = *(const bf16x8*)(ar1 + k0 + 8 + 16 * half);
            fb0.h[0] = ((const bf16x8*)(brow0 + k0))[0];
            fb0.h[1] = ((const bf16x8*)(brow0 + k0))[1];
            fb1.h[0] = ((const bf16x8*)(brow1 + k0))[0];
            fb1.h[1] = ((const bf16x8*)(brow1 + k0))[1];
            acc00 = wmma_bf16(fa0.v, fb0.v, acc00);
            acc01 = wmma_bf16(fa0.v, fb1.v, acc01);
            acc10 = wmma_bf16(fa1.v, fb0.v, acc10);
            acc11 = wmma_bf16(fa1.v, fb1.v, acc11);
        }
#pragma unroll
        for (int r = 0; r < 8; ++r) {
            int M0 = r + 8 * half, M1 = 16 + r + 8 * half;
            Hls[M0 * H_STR + n0 +      nlo] = (bf16)gelu_exact(acc00[r]);
            Hls[M0 * H_STR + n0 + 16 + nlo] = (bf16)gelu_exact(acc01[r]);
            Hls[M1 * H_STR + n0 +      nlo] = (bf16)gelu_exact(acc10[r]);
            Hls[M1 * H_STR + n0 + 16 + nlo] = (bf16)gelu_exact(acc11[r]);
        }
    }
    __syncthreads();

    // ---- phase 2: H @ W2 -> ok (unweighted; gates applied in combine) ----
    const bf16* w2e = w2t + (size_t)e * HID * FF;
    const bf16* hr0 = &Hls[nlo * H_STR];
    const bf16* hr1 = &Hls[(16 + nlo) * H_STR];
#pragma unroll 1
    for (int np = wave; np < HID / 32; np += 8) {      // 32 N-pairs, 4 per wave
        const int n0 = np * 32;
        v8f acc00 = {0,0,0,0,0,0,0,0}, acc01 = {0,0,0,0,0,0,0,0};
        v8f acc10 = {0,0,0,0,0,0,0,0}, acc11 = {0,0,0,0,0,0,0,0};
        const bf16* brow0 = w2e + (size_t)(n0 + nlo) * FF + 16 * half;
        const bf16* brow1 = brow0 + (size_t)16 * FF;
#pragma unroll 1
        for (int k0 = 0; k0 < FF; k0 += 32) {
            FragU fa0, fa1, fb0, fb1;
            fa0.h[0] = *(const bf16x8*)(hr0 + k0 +     16 * half);
            fa0.h[1] = *(const bf16x8*)(hr0 + k0 + 8 + 16 * half);
            fa1.h[0] = *(const bf16x8*)(hr1 + k0 +     16 * half);
            fa1.h[1] = *(const bf16x8*)(hr1 + k0 + 8 + 16 * half);
            fb0.h[0] = ((const bf16x8*)(brow0 + k0))[0];
            fb0.h[1] = ((const bf16x8*)(brow0 + k0))[1];
            fb1.h[0] = ((const bf16x8*)(brow1 + k0))[0];
            fb1.h[1] = ((const bf16x8*)(brow1 + k0))[1];
            acc00 = wmma_bf16(fa0.v, fb0.v, acc00);
            acc01 = wmma_bf16(fa0.v, fb1.v, acc01);
            acc10 = wmma_bf16(fa1.v, fb0.v, acc10);
            acc11 = wmma_bf16(fa1.v, fb1.v, acc11);
        }
#pragma unroll
        for (int r = 0; r < 8; ++r) {
            int M0 = r + 8 * half, M1 = 16 + r + 8 * half;
            if (base + M0 < cnt) {
                float* orow = ok + (size_t)aid[M0] * HID;
                orow[n0 +      nlo] = acc00[r];
                orow[n0 + 16 + nlo] = acc01[r];
            }
            if (base + M1 < cnt) {
                float* orow = ok + (size_t)aid[M1] * HID;
                orow[n0 +      nlo] = acc10[r];
                orow[n0 + 16 + nlo] = acc11[r];
            }
        }
    }
}

// ---------------- kernel 6: gated combine (deterministic, no atomics) ----------------
__global__ __launch_bounds__(256) void combine_kernel(
    const float* __restrict__ ok, const float* __restrict__ tw, float* __restrict__ out)
{
    size_t i = (size_t)blockIdx.x * 256 + threadIdx.x;   // over NTOK*HID
    int t = (int)(i >> 10);
    int h = (int)(i & (HID - 1));
    out[i] = tw[t * 2 + 0] * ok[((size_t)t * 2 + 0) * HID + h]
           + tw[t * 2 + 1] * ok[((size_t)t * 2 + 1) * HID + h];
}

// ---------------- host launcher ----------------
static inline size_t align256(size_t x) { return (x + 255) & ~(size_t)255; }

extern "C" void kernel_launch(void* const* d_in, const int* in_sizes, int n_in,
                              void* d_out, int out_size, void* d_ws, size_t ws_size,
                              hipStream_t stream) {
    (void)in_sizes; (void)n_in; (void)out_size; (void)ws_size;
    const float* x   = (const float*)d_in[0];   // [4,2048,1024]
    const float* rw  = (const float*)d_in[1];   // [8,1024]
    const float* w1  = (const float*)d_in[2];   // [8,1024,2048]
    const float* w2  = (const float*)d_in[3];   // [8,2048,1024]
    float* out = (float*)d_out;

    // workspace carve (~151 MB)
    char* p = (char*)d_ws;
    bf16* xb    = (bf16*)p;  p += align256((size_t)NTOK * HID * sizeof(bf16));
    bf16* w1t   = (bf16*)p;  p += align256((size_t)NEXP * FF * HID * sizeof(bf16));
    bf16* w2t   = (bf16*)p;  p += align256((size_t)NEXP * HID * FF * sizeof(bf16));
    float* ok   = (float*)p; p += align256((size_t)NTOK * 2 * HID * sizeof(float));
    int*  tidx  = (int*)p;   p += align256((size_t)NTOK * 2 * sizeof(int));
    float* tw   = (float*)p; p += align256((size_t)NTOK * 2 * sizeof(float));
    int*  counts= (int*)p;   p += align256((size_t)NEXP * sizeof(int));
    int*  lists = (int*)p;   p += align256((size_t)NEXP * NTOK * sizeof(int));

    router_kernel<<<NTOK / 8, 256, 0, stream>>>(x, rw, tidx, tw);
    cvt_x_kernel<<<(NTOK * HID) / 256, 256, 0, stream>>>(x, xb, NTOK * HID);
    transpose_cvt_kernel<<<dim3(FF / 32, HID / 32, NEXP), 256, 0, stream>>>(w1, w1t, HID, FF);
    transpose_cvt_kernel<<<dim3(HID / 32, FF / 32, NEXP), 256, 0, stream>>>(w2, w2t, FF, HID);
    zero_counts_kernel<<<1, 32, 0, stream>>>(counts);
    scatter_kernel<<<NTOK / 256, 256, 0, stream>>>(tidx, counts, lists);
    moe_ffn_kernel<<<NEXP * 256, 256, 0, stream>>>(xb, w1t, w2t, counts, lists, ok);
    combine_kernel<<<(NTOK * HID) / 256, 256, 0, stream>>>(ok, tw, out);
}
